// MedPoseAttention_29807073034705
// MI455X (gfx1250) — matrix operand order repaired and checked by reading, not compiled, not run on hardware
//
#include <hip/hip_runtime.h>
#include <hip/hip_bf16.h>
#include <stdint.h>

// MI455X / gfx1250: wave32, WMMA 16x16x32 f16 -> f32, async global->LDS DMA.
typedef __attribute__((ext_vector_type(16))) _Float16 v16h;
typedef __attribute__((ext_vector_type(8)))  float    v8f;

#define LQ 1024
#define DMODEL 1024
#define NROWS 4096   // B * LQ
#define HD 64

__device__ __forceinline__ v8f wmma_f16(v16h a, v16h b, v8f c) {
  // (neg_a, A, neg_b, B, c_mod, C, reuse_a, reuse_b)
  return __builtin_amdgcn_wmma_f32_16x16x32_f16(false, a, false, b, (short)0, c,
                                                false, false);
}

// A-fragment slot -> K offset within a 16x32 f16 tile (contiguous: k = 16*hi + t, matches B).
__device__ __forceinline__ int a_slot_k(int t, int hi) {
  return t + (hi << 4);
}

// CDNA5 async global->LDS copy, 16B per lane, tracked by ASYNCcnt.
// LDS address = wave-relative byte offset = low 32 bits of the flat pointer
// (flat->LDS aperture mapping uses addr[31:0], ISA Sec. 10.2).
__device__ __forceinline__ void async_ld_b128(const _Float16* g, _Float16* l) {
  asm volatile("global_load_async_to_lds_b128 %0, %1, off"
               :: "v"((unsigned)(uintptr_t)l), "v"(g)
               : "memory");
}
__device__ __forceinline__ void wait_async0() {
  asm volatile("s_wait_asynccnt 0x0" ::: "memory");
}

// ---------------------------------------------------------------------------
// Kernel 1: fused per-head QKV projection.  X[4096,1024](f32) @ Wstacked + b
// -> f16 buffer [row, h*64+d].  z==0 additionally writes f32 residual (== q).
// Wq/Wk/Wv flat layout: (h, d, m) at h*65536 + d*64 + m; stacked col c=h*64+m.
// ---------------------------------------------------------------------------
__global__ __launch_bounds__(256) void proj_kernel(
    const float* __restrict__ queries, const float* __restrict__ context,
    const float* __restrict__ Wq, const float* __restrict__ bq,
    const float* __restrict__ Wk, const float* __restrict__ bk,
    const float* __restrict__ Wv, const float* __restrict__ bv,
    _Float16* __restrict__ qbuf, _Float16* __restrict__ kbuf,
    _Float16* __restrict__ vbuf, float* __restrict__ residual) {
  const int which = blockIdx.z;
  const float* X;  const float* W;  const float* bias;  _Float16* dst;
  if (which == 0)      { X = queries; W = Wq; bias = bq; dst = qbuf; }
  else if (which == 1) { X = context; W = Wk; bias = bk; dst = kbuf; }
  else                 { X = context; W = Wv; bias = bv; dst = vbuf; }

  const int lane = threadIdx.x & 31;
  const int wid  = threadIdx.x >> 5;
  const int row  = lane & 15;       // M (A/C) or N (B) sub-index
  const int hi   = lane >> 4;       // half-wave select

  const int m0 = blockIdx.x * 16;                   // output row tile
  const int n0 = (blockIdx.y * 8 + wid) * 16;       // output col tile
  const int h  = n0 >> 6;                           // head (16 cols stay in one head)
  const int nIn = n0 & 63;

  v8f acc = {};
  for (int k0 = 0; k0 < DMODEL; k0 += 32) {
    v16h a, bf;
#pragma unroll
    for (int t = 0; t < 16; ++t) {
      const int k = k0 + a_slot_k(t, hi);
      a[t] = (_Float16)X[(size_t)(m0 + row) * DMODEL + k];
    }
#pragma unroll
    for (int t = 0; t < 16; ++t) {
      const int kk = k0 + (hi << 4) + t;            // B: lane=N, K contiguous per half
      bf[t] = (_Float16)W[(size_t)h * 65536 + (size_t)kk * HD + nIn + row];
    }
    acc = wmma_f16(a, bf, acc);
  }

  const float bb = bias[n0 + row];                  // bias flat over [H*HD]
#pragma unroll
  for (int r = 0; r < 8; ++r) {
    const int gm = m0 + r + (hi << 3);              // C: row = r + 8*hi
    dst[(size_t)gm * DMODEL + n0 + row] = (_Float16)(acc[r] + bb);
  }
  if (which == 0) {
#pragma unroll
    for (int r = 0; r < 8; ++r) {
      const int gm = m0 + r + (hi << 3);
      residual[(size_t)gm * DMODEL + n0 + row] = acc[r] + bb;
    }
  }
}

// ---------------------------------------------------------------------------
// Kernel 2: flash attention.  Block = 4 waves sharing one (b,h); each wave
// owns 16 query rows.  K/V 32x64 tiles are staged once per block into LDS by
// async global->LDS b128 DMA (double-buffered, overlapped with WMMA), then
// per KV block: 4 WMMAs for S = Q K^T, online softmax with 16-lane shfl
// reductions, P bounced through LDS (C-layout -> A-layout), 4 WMMAs O += P V.
// ---------------------------------------------------------------------------
__global__ __launch_bounds__(128) void attn_kernel(
    const _Float16* __restrict__ qbuf, const _Float16* __restrict__ kbuf,
    const _Float16* __restrict__ vbuf, _Float16* __restrict__ multi) {
  __shared__ _Float16 ktile[2][32 * 64];   // 2 x 4KB
  __shared__ _Float16 vtile[2][32 * 64];   // 2 x 4KB
  __shared__ _Float16 pshare[4][16 * 32];  // 4 x 1KB

  const int tid  = threadIdx.x;
  const int lane = tid & 31;
  const int wid  = tid >> 5;
  const int row  = lane & 15;
  const int hi   = lane >> 4;

  const int bh = blockIdx.x;                 // 0..63
  const int b  = bh >> 4, h = bh & 15;
  const int q0 = (blockIdx.y * 4 + wid) * 16;

  // base of (b, l=0, head h) in the [row, h*64+d] buffers
  const size_t base = (size_t)b * LQ * DMODEL + (size_t)h * HD;
  const _Float16* kg = kbuf + base;
  const _Float16* vg = vbuf + base;

  // Cooperative async stage of one 32x64 K tile + V tile (4KB each):
  // 256 16B chunks per tile, 128 threads -> 2 chunks each per tile.
  auto stage_kv = [&](int buf, int j0) {
#pragma unroll
    for (int i = 0; i < 2; ++i) {
      const int c   = tid + (i << 7);     // chunk id 0..255
      const int r   = c >> 3;             // kv row within block
      const int seg = (c & 7) << 3;       // halves within row
      async_ld_b128(kg + (size_t)(j0 + r) * DMODEL + seg, &ktile[buf][r * 64 + seg]);
      async_ld_b128(vg + (size_t)(j0 + r) * DMODEL + seg, &vtile[buf][r * 64 + seg]);
    }
  };

  // Q A-fragments for the two K-chunks (d 0..31, 32..63) -- loaded once.
  v16h aq[2];
#pragma unroll
  for (int c = 0; c < 2; ++c)
#pragma unroll
    for (int t = 0; t < 16; ++t)
      aq[c][t] = qbuf[base + (size_t)(q0 + row) * DMODEL + c * 32 + a_slot_k(t, hi)];

  float mrow[8], lrow[8];
#pragma unroll
  for (int r = 0; r < 8; ++r) { mrow[r] = -3.0e38f; lrow[r] = 0.0f; }
  v8f o[4] = {{}, {}, {}, {}};

  _Float16* P = pshare[wid];
  const float scale = 0.125f;  // 1/sqrt(64)

  // Prologue: stage first KV block.
  stage_kv(0, 0);
  wait_async0();
  __syncthreads();

  for (int j0 = 0; j0 < LQ; j0 += 32) {
    const int cur = (j0 >> 5) & 1;
    if (j0 + 32 < LQ) stage_kv(cur ^ 1, j0 + 32);          // overlap DMA w/ WMMA
    if (j0 + 64 < LQ) {                                    // global_prefetch_b8
      __builtin_prefetch(kg + (size_t)(j0 + 64 + row) * DMODEL, 0, 0);
      __builtin_prefetch(vg + (size_t)(j0 + 64 + row) * DMODEL, 0, 0);
    }
    const _Float16* kt = ktile[cur];
    const _Float16* vt = vtile[cur];

    // S tiles: cols j0..j0+15 (s0) and j0+16..j0+31 (s1); K read from LDS.
    v8f s0 = {}, s1 = {};
#pragma unroll
    for (int c = 0; c < 2; ++c) {
      v16h b0, b1;
#pragma unroll
      for (int t = 0; t < 16; ++t) {
        const int d = c * 32 + (hi << 4) + t;  // B[kk][n] = K[j0+n][d]
        b0[t] = kt[row * 64 + d];
        b1[t] = kt[(16 + row) * 64 + d];
      }
      s0 = wmma_f16(aq[c], b0, s0);
      s1 = wmma_f16(aq[c], b1, s1);
    }

    float pv0[8], pv1[8], alpha[8];
#pragma unroll
    for (int r = 0; r < 8; ++r) {
      float v0 = s0[r] * scale, v1 = s1[r] * scale;
      float mx = fmaxf(v0, v1);
#pragma unroll
      for (int msk = 8; msk >= 1; msk >>= 1) mx = fmaxf(mx, __shfl_xor(mx, msk, 32));
      const float mnew = fmaxf(mrow[r], mx);
      alpha[r] = __expf(mrow[r] - mnew);
      mrow[r] = mnew;
      pv0[r] = __expf(v0 - mnew);
      pv1[r] = __expf(v1 - mnew);
      float rs = pv0[r] + pv1[r];
#pragma unroll
      for (int msk = 8; msk >= 1; msk >>= 1) rs += __shfl_xor(rs, msk, 32);
      lrow[r] = lrow[r] * alpha[r] + rs;
    }
#pragma unroll
    for (int n = 0; n < 4; ++n)
#pragma unroll
      for (int r = 0; r < 8; ++r) o[n][r] *= alpha[r];

    // P: C-layout -> row-major LDS [16 rows q][32 cols kv] -> A-layout reload.
#pragma unroll
    for (int r = 0; r < 8; ++r) {
      P[(r + (hi << 3)) * 32 + row]      = (_Float16)pv0[r];
      P[(r + (hi << 3)) * 32 + 16 + row] = (_Float16)pv1[r];
    }
    v16h pa;
#pragma unroll
    for (int t = 0; t < 16; ++t) pa[t] = P[row * 32 + a_slot_k(t, hi)];

    // O += P * V  (V B-fragments from LDS: lane = d-col, K = kv row in block)
#pragma unroll
    for (int n = 0; n < 4; ++n) {
      v16h bvf;
#pragma unroll
      for (int t = 0; t < 16; ++t) {
        const int kk = (hi << 4) + t;
        bvf[t] = vt[kk * 64 + n * 16 + row];
      }
      o[n] = wmma_f16(pa, bvf, o[n]);
    }

    // Next tile's DMA must have landed before anyone reads it; and all waves
    // must be done reading `cur` before iteration j0+32 overwrites it.
    wait_async0();
    __syncthreads();
  }

#pragma unroll
  for (int r = 0; r < 8; ++r) {
    const float inv = 1.0f / lrow[r];
    const int gm = q0 + r + (hi << 3);
#pragma unroll
    for (int n = 0; n < 4; ++n)
      multi[base + (size_t)gm * DMODEL + n * 16 + row] = (_Float16)(o[n][r] * inv);
  }
}

// ---------------------------------------------------------------------------
// Kernel 3: out = multi[4096,1024](f16) @ Wo[1024,1024](f32) + bo -> f32.
// ---------------------------------------------------------------------------
__global__ __launch_bounds__(256) void outproj_kernel(
    const _Float16* __restrict__ multi, const float* __restrict__ Wo,
    const float* __restrict__ bo, float* __restrict__ out) {
  const int lane = threadIdx.x & 31;
  const int wid  = threadIdx.x >> 5;
  const int row  = lane & 15;
  const int hi   = lane >> 4;

  const int m0 = blockIdx.x * 16;
  const int n0 = (blockIdx.y * 8 + wid) * 16;

  v8f acc = {};
  for (int k0 = 0; k0 < DMODEL; k0 += 32) {
    v16h a, bf;
#pragma unroll
    for (int t = 0; t < 16; ++t)
      a[t] = multi[(size_t)(m0 + row) * DMODEL + k0 + a_slot_k(t, hi)];
#pragma unroll
    for (int t = 0; t < 16; ++t) {
      const int kk = k0 + (hi << 4) + t;
      bf[t] = (_Float16)Wo[(size_t)kk * DMODEL + n0 + row];
    }
    acc = wmma_f16(a, bf, acc);
  }
  const float bb = bo[n0 + row];
#pragma unroll
  for (int r = 0; r < 8; ++r)
    out[(size_t)(m0 + r + (hi << 3)) * DMODEL + n0 + row] = acc[r] + bb;
}

// ---------------------------------------------------------------------------
extern "C" void kernel_launch(void* const* d_in, const int* in_sizes, int n_in,
                              void* d_out, int out_size, void* d_ws, size_t ws_size,
                              hipStream_t stream) {
  const float* queries = (const float*)d_in[0];
  const float* context = (const float*)d_in[1];
  const float* Wq = (const float*)d_in[2];
  const float* bq = (const float*)d_in[3];
  const float* Wk = (const float*)d_in[4];
  const float* bk = (const float*)d_in[5];
  const float* Wv = (const float*)d_in[6];
  const float* bv = (const float*)d_in[7];
  const float* Wo = (const float*)d_in[8];
  const float* bo = (const float*)d_in[9];

  float* out      = (float*)d_out;                       // [4,1024,1024]
  float* residual = out + (size_t)NROWS * DMODEL;        // [4,1024,1024]

  _Float16* qbuf  = (_Float16*)d_ws;                     // 8 MB each
  _Float16* kbuf  = qbuf + (size_t)NROWS * DMODEL;
  _Float16* vbuf  = kbuf + (size_t)NROWS * DMODEL;
  _Float16* multi = vbuf + (size_t)NROWS * DMODEL;

  // 1) QKV projections (z: 0=q, 1=k, 2=v); q pass also emits f32 residual.
  proj_kernel<<<dim3(NROWS / 16, 8, 3), 256, 0, stream>>>(
      queries, context, Wq, bq, Wk, bk, Wv, bv, qbuf, kbuf, vbuf, residual);

  // 2) Flash attention per (b,h); 4 waves/block share async-staged KV tiles.
  attn_kernel<<<dim3(64, 16), 128, 0, stream>>>(qbuf, kbuf, vbuf, multi);

  // 3) Output projection.
  outproj_kernel<<<dim3(NROWS / 16, 8), 256, 0, stream>>>(multi, Wo, bo, out);
}
